// ContextBERTSelfAttention_77541339562306
// MI455X (gfx1250) — matrix operand order repaired and checked by reading, not compiled, not run on hardware
//
#include <hip/hip_runtime.h>
#include <hip/hip_bf16.h>

#define B_  2
#define S_  2048
#define H_  768
#define NH_ 12
#define DH_ 64

typedef __bf16 bf16;
typedef __attribute__((ext_vector_type(16))) __bf16 v16bf;
typedef __attribute__((ext_vector_type(8)))  __bf16 v8bf;
typedef __attribute__((ext_vector_type(8)))  float  v8f;
typedef __attribute__((ext_vector_type(4)))  float  v4f;

#define WMMA_BF16(a, b, c) \
  __builtin_amdgcn_wmma_f32_16x16x32_bf16(false, (a), false, (b), (short)0, (c), false, false)

// ---------- WMMA fragment helpers (wave32 layouts, ISA 7.12.2) ----------
// A tile 16(M)x32(K) from row-major bf16: two contiguous 16B runs per lane.
__device__ inline v16bf a_frag_bf16(const bf16* base, int ld) {
  const int L = threadIdx.x & 31;
  const int m = L & 15;
  const int kb = (L < 16) ? 0 : 8;
  const bf16* row = base + (size_t)m * ld;
  v8bf lo = *(const v8bf*)(row + kb);       // K = kb .. kb+7
  v8bf hi = *(const v8bf*)(row + 16 + kb);  // K = 16+kb .. 16+kb+7
  v16bf a;
#pragma unroll
  for (int e = 0; e < 8; ++e) { a[e] = lo[e]; a[e + 8] = hi[e]; }
  return a;
}
// A tile 16(M)x32(K) from row-major fp32: 4x float4 loads + cvt per lane.
__device__ inline v16bf a_frag_f32(const float* base, int ld) {
  const int L = threadIdx.x & 31;
  const int m = L & 15;
  const int kb = (L < 16) ? 0 : 8;
  const float* row = base + (size_t)m * ld;
  v4f l0 = *(const v4f*)(row + kb);
  v4f l1 = *(const v4f*)(row + kb + 4);
  v4f h0 = *(const v4f*)(row + 16 + kb);
  v4f h1 = *(const v4f*)(row + 16 + kb + 4);
  v16bf a;
#pragma unroll
  for (int e = 0; e < 4; ++e) {
    a[e]      = (bf16)l0[e];
    a[e + 4]  = (bf16)l1[e];
    a[e + 8]  = (bf16)h0[e];
    a[e + 12] = (bf16)h1[e];
  }
  return a;
}
// B tile 32(K)x16(N) from row-major fp32 base[k*ld + n] (coalesced across lanes).
__device__ inline v16bf b_frag_f32(const float* base, int ld) {
  const int L = threadIdx.x & 31;
  const int n = L & 15;
  const int kb = (L < 16) ? 0 : 8;
  v16bf b;
#pragma unroll
  for (int e = 0; e < 16; ++e)
    b[e] = (bf16)base[(size_t)(kb + (e < 8 ? e : e + 8)) * ld + n];
  return b;
}
// B tile for A@srcT: logical B[k][n] = src[n*ld + k]. Per lane: two 16B runs (K = kb.., 16+kb..).
__device__ inline v16bf bT_frag_bf16(const bf16* src, int ld) {
  const int L = threadIdx.x & 31;
  const int n = L & 15;
  const int kb = (L < 16) ? 0 : 8;
  const bf16* p = src + (size_t)n * ld + kb;
  v8bf lo = *(const v8bf*)(p);
  v8bf hi = *(const v8bf*)(p + 16);
  v16bf b;
#pragma unroll
  for (int e = 0; e < 8; ++e) { b[e] = lo[e]; b[e + 8] = hi[e]; }
  return b;
}
// C/D element r -> (row, col) for this lane
__device__ inline void cd_coord(int r, int& row, int& col) {
  const int L = threadIdx.x & 31;
  row = (L < 16) ? r : (r + 8);
  col = L & 15;
}

// ---------- Kernel 1: fused Q/K/V projection (3 matrices per block) ----------
// grid(256, 48), block(32). A-frag loaded once per k-step, feeds 3 WMMAs.
// q,k fp32 head-layout; v bf16 TRANSPOSED [B,NH,DH,S].
__global__ void k_qkv(const float* __restrict__ hidden,
                      const float* __restrict__ Wq, const float* __restrict__ bq,
                      const float* __restrict__ Wk, const float* __restrict__ bk,
                      const float* __restrict__ Wv, const float* __restrict__ bv,
                      float* __restrict__ qf, float* __restrict__ kf,
                      bf16* __restrict__ vt) {
  const int m0 = blockIdx.x * 16;
  const int n0 = blockIdx.y * 16;
  v8f accq = {}, acck = {}, accv = {};
  for (int k0 = 0; k0 < H_; k0 += 32) {
    v16bf a = a_frag_f32(hidden + (size_t)m0 * H_ + k0, H_);
    v16bf b0 = b_frag_f32(Wq + (size_t)k0 * H_ + n0, H_);
    v16bf b1 = b_frag_f32(Wk + (size_t)k0 * H_ + n0, H_);
    v16bf b2 = b_frag_f32(Wv + (size_t)k0 * H_ + n0, H_);
    accq = WMMA_BF16(a, b0, accq);
    acck = WMMA_BF16(a, b1, acck);
    accv = WMMA_BF16(a, b2, accv);
  }
#pragma unroll
  for (int r = 0; r < 8; ++r) {
    int row, col; cd_coord(r, row, col);
    const int m = m0 + row, n = n0 + col;
    const int bb = m / S_, s = m % S_;
    const int h = n / DH_, d = n % DH_;
    const size_t hidx = (((size_t)bb * NH_ + h) * S_ + s) * DH_ + d;
    qf[hidx] = accq[r] + bq[n];
    kf[hidx] = acck[r] + bk[n];
    vt[(((size_t)bb * NH_ + h) * DH_ + d) * S_ + s] = (bf16)(accv[r] + bv[n]);
  }
}

// ---------- Kernel 2: [M,768]x[768,64]+bias, both weights + all n fused ----------
// grid(M/16), block(32). A-frag loaded once per k-step, feeds 8 WMMAs:
// side(2) x ntile(4). Streams the A tensor from HBM exactly once.
__global__ void k_proj64(const float* __restrict__ A,
                         const float* __restrict__ Wa, const float* __restrict__ ba,
                         const float* __restrict__ Wb, const float* __restrict__ bb_,
                         float* __restrict__ outa, float* __restrict__ outb) {
  const int m0 = blockIdx.x * 16;
  v8f acc[2][4] = {};
  for (int k0 = 0; k0 < H_; k0 += 32) {
    v16bf a = a_frag_f32(A + (size_t)m0 * H_ + k0, H_);
#pragma unroll
    for (int nt = 0; nt < 4; ++nt) {
      v16bf b0 = b_frag_f32(Wa + (size_t)k0 * DH_ + nt * 16, DH_);
      v16bf b1 = b_frag_f32(Wb + (size_t)k0 * DH_ + nt * 16, DH_);
      acc[0][nt] = WMMA_BF16(a, b0, acc[0][nt]);
      acc[1][nt] = WMMA_BF16(a, b1, acc[1][nt]);
    }
  }
#pragma unroll
  for (int nt = 0; nt < 4; ++nt) {
#pragma unroll
    for (int r = 0; r < 8; ++r) {
      int row, col; cd_coord(r, row, col);
      const int n = nt * 16 + col;
      outa[(size_t)(m0 + row) * DH_ + n] = acc[0][nt][r] + ba[n];
      outb[(size_t)(m0 + row) * DH_ + n] = acc[1][nt][r] + bb_[n];
    }
  }
}

// ---------- Kernel 3: concat-GEMM (K=128) + tanh lambda gate + blend ----------
__global__ void k_gate(const float* __restrict__ hcq, const float* __restrict__ ceq,
                       const float* __restrict__ hck, const float* __restrict__ cek,
                       const float* __restrict__ Whcq, const float* __restrict__ bhcq,
                       const float* __restrict__ Whck, const float* __restrict__ bhck,
                       const float* __restrict__ wlqc, const float* __restrict__ wlqq,
                       const float* __restrict__ wlkc, const float* __restrict__ wlkk,
                       const float* __restrict__ qf, const float* __restrict__ kf,
                       bf16* __restrict__ cq_layer, bf16* __restrict__ ck_layer) {
  __shared__ bf16 catile[16][128];
  __shared__ float ctile[16][DH_ + 4];
  const int m0 = blockIdx.x * 16;
  const int L = threadIdx.x;
  for (int side = 0; side < 2; ++side) {
    const float* hc = side ? hck : hcq;
    const float* ce = side ? cek : ceq;
    const float* Wm = side ? Whck : Whcq;
    const float* bm = side ? bhck : bhcq;
    const float* wlc = side ? wlkc : wlqc;
    const float* wlx = side ? wlkk : wlqq;
    const float* base = side ? kf : qf;
    bf16* outp = side ? ck_layer : cq_layer;
    // stage concat([hc, broadcast(ce)]) tile -> LDS bf16 [16][128]
    for (int t = L; t < 16 * 128; t += 32) {
      const int r = t >> 7, c = t & 127;
      const size_t m = (size_t)m0 + r;
      float v;
      if (c < DH_) {
        v = hc[m * DH_ + c];
      } else {
        const size_t bb = m / ((size_t)NH_ * S_);
        const size_t s  = m % S_;
        v = ce[(bb * S_ + s) * DH_ + (c - DH_)];
      }
      catile[r][c] = (bf16)v;
    }
    __syncthreads();
    for (int nt = 0; nt < 4; ++nt) {
      v8f acc = {};
      for (int k0 = 0; k0 < 128; k0 += 32) {
        v16bf a = a_frag_bf16(&catile[0][k0], 128);
        v16bf b = b_frag_f32(Wm + (size_t)k0 * DH_ + nt * 16, DH_);
        acc = WMMA_BF16(a, b, acc);
      }
#pragma unroll
      for (int r = 0; r < 8; ++r) {
        int row, col; cd_coord(r, row, col);
        ctile[row][nt * 16 + col] = acc[r] + bm[nt * 16 + col];
      }
    }
    __syncthreads();
    if (L < 16) {
      const size_t m = (size_t)m0 + L;
      float accl = 0.f;
      for (int d = 0; d < DH_; ++d)
        accl += ctile[L][d] * wlc[d] + base[m * DH_ + d] * wlx[d];
      const float lam = tanhf(accl);
      for (int d = 0; d < DH_; ++d) {
        const float o = (1.f - lam) * base[m * DH_ + d] + lam * ctile[L][d];
        outp[m * DH_ + d] = (bf16)o;
      }
    }
    __syncthreads();
  }
}

// ---------- Kernel 4: fused scores -> softmax -> probs -> P@V -> ctx ----------
// grid(S/16, B*NH), block(256) = 8 waves.
// LDS: P[16][2052] f32 | Cpart[2][4][16][16] f32 | Qs[16][64] bf16 (async-staged)
__global__ void __launch_bounds__(256) k_attn(
    const bf16* __restrict__ cq_layer, const bf16* __restrict__ ck_layer,
    const bf16* __restrict__ vt, const float* __restrict__ mask,
    float* __restrict__ ctx_out, float* __restrict__ probs_out) {
  constexpr int PLD = S_ + 4;  // stride mod 64 banks == 4 -> conflict-free rows
  constexpr unsigned CPART_OFF = (unsigned)(16 * PLD * sizeof(float));
  constexpr unsigned QS_OFF    = CPART_OFF + 2 * 4 * 16 * 16 * sizeof(float);
  extern __shared__ char smem[];
  float* P     = (float*)smem;
  float* Cpart = (float*)(smem + CPART_OFF);
  bf16*  Qs    = (bf16*)(smem + QS_OFF);
  const int qt = blockIdx.x;
  const int bh = blockIdx.y;
  const int bb = bh / NH_;
  const int hh = bh % NH_;
  const int wid = threadIdx.x >> 5;
  const int L = threadIdx.x & 31;
  const bf16* Q  = cq_layer + ((size_t)bh * S_ + (size_t)qt * 16) * DH_;
  const bf16* K  = ck_layer + (size_t)bh * S_ * DH_;
  const bf16* Vt = vt + (size_t)bh * DH_ * S_;  // [DH][S]
  const float scale = 0.125f;  // 1/sqrt(DH)

  // --- async-stage Q tile (16x64 bf16 = 2KB) into LDS, ASYNCcnt-tracked ---
  if (threadIdx.x < 128) {
    const unsigned lds_off = QS_OFF + threadIdx.x * 16u;
    const bf16* gp = Q + threadIdx.x * 8;  // 8 bf16 = 16B per thread
    asm volatile("global_load_async_to_lds_b128 %0, %1, off"
                 :: "v"(lds_off), "v"(gp) : "memory");
  }
  asm volatile("s_wait_asynccnt 0" ::: "memory");
  __syncthreads();

  // --- scores: Q(16xDH) @ K^T, waves split the 2048 key columns ---
  v16bf aq0 = a_frag_bf16(Qs, DH_);        // d = 0..31
  v16bf aq1 = a_frag_bf16(Qs + 32, DH_);   // d = 32..63
  for (int kt = wid * 16; kt < wid * 16 + 16; ++kt) {
    const bf16* Kt = K + (size_t)kt * 16 * DH_;
    __builtin_prefetch(Kt + 16 * DH_ + ((size_t)(L & 15)) * DH_, 0, 1);  // next tile
    v16bf b0 = bT_frag_bf16(Kt, DH_);
    v16bf b1 = bT_frag_bf16(Kt + 32, DH_);
    v8f sc = {};
    sc = WMMA_BF16(aq0, b0, sc);
    sc = WMMA_BF16(aq1, b1, sc);
#pragma unroll
    for (int r = 0; r < 8; ++r) {
      int row, col; cd_coord(r, row, col);
      const int kc = kt * 16 + col;
      P[row * PLD + kc] = sc[r] * scale + mask[(size_t)bb * S_ + kc];
    }
  }
  __syncthreads();

  // --- softmax: 2 rows per wave, shuffle reductions; write probs to d_out ---
  for (int rr = 0; rr < 2; ++rr) {
    const int r = wid * 2 + rr;
    float* row = P + r * PLD;
    float mx = -3.0e38f;
    for (int c = L; c < S_; c += 32) mx = fmaxf(mx, row[c]);
    for (int o = 16; o; o >>= 1) mx = fmaxf(mx, __shfl_xor(mx, o, 32));
    float sum = 0.f;
    for (int c = L; c < S_; c += 32) {
      const float e = __expf(row[c] - mx);
      row[c] = e;
      sum += e;
    }
    for (int o = 16; o; o >>= 1) sum += __shfl_xor(sum, o, 32);
    const float inv = 1.f / sum;
    const size_t prow = ((size_t)bh * S_ + qt * 16 + r) * S_;
    for (int c = L; c < S_; c += 32) {
      const float p = row[c] * inv;
      row[c] = p;
      probs_out[prow + c] = p;
    }
  }
  __syncthreads();

  // --- P @ V: wave w -> ntile = w&3, k-half = w>>2; V read from Vt [DH][S] ---
  {
    const int nt = wid & 3, kh = wid >> 2;
    v8f acc = {};
    for (int k0 = kh * 1024; k0 < kh * 1024 + 1024; k0 += 32) {
      // A from LDS P (fp32 -> bf16), two contiguous 8-float runs per lane
      v16bf a;
      {
        const int m = L & 15, kb = (L < 16) ? 0 : 8;
        const float* prow = P + m * PLD + k0;
        v4f l0 = *(const v4f*)(prow + kb);
        v4f l1 = *(const v4f*)(prow + kb + 4);
        v4f h0 = *(const v4f*)(prow + 16 + kb);
        v4f h1 = *(const v4f*)(prow + 16 + kb + 4);
#pragma unroll
        for (int e = 0; e < 4; ++e) {
          a[e]      = (bf16)l0[e];
          a[e + 4]  = (bf16)l1[e];
          a[e + 8]  = (bf16)h0[e];
          a[e + 12] = (bf16)h1[e];
        }
      }
      // B[k][n] = Vt[n][k]: two 16B runs per lane
      v16bf b = bT_frag_bf16(Vt + (size_t)(nt * 16) * S_ + k0, S_);
      acc = WMMA_BF16(a, b, acc);
    }
    float* cp = Cpart + ((kh * 4 + nt) * 16) * 16;
#pragma unroll
    for (int r = 0; r < 8; ++r) {
      int row, col; cd_coord(r, row, col);
      cp[row * 16 + col] = acc[r];
    }
  }
  __syncthreads();

  // --- reduce k-halves, write ctx in [B,S,H] layout ---
  if (wid < 4) {
    const int nt = wid;
    const float* c0 = Cpart + ((0 * 4 + nt) * 16) * 16;
    const float* c1 = Cpart + ((1 * 4 + nt) * 16) * 16;
    for (int t = L; t < 256; t += 32) {
      const int row = t >> 4, col = t & 15;
      const float val = c0[t] + c1[t];
      const size_t s = (size_t)qt * 16 + row;
      ctx_out[((size_t)bb * S_ + s) * H_ + hh * DH_ + nt * 16 + col] = val;
    }
  }
}

// ---------- host launcher ----------
extern "C" void kernel_launch(void* const* d_in, const int* in_sizes, int n_in,
                              void* d_out, int out_size, void* d_ws, size_t ws_size,
                              hipStream_t stream) {
  (void)in_sizes; (void)n_in; (void)out_size; (void)ws_size;
  const float* hidden = (const float*)d_in[0];
  const float* mask   = (const float*)d_in[1];
  const float* ctxemb = (const float*)d_in[2];
  const float* hctx   = (const float*)d_in[3];
  const float* Wq = (const float*)d_in[4];  const float* bq = (const float*)d_in[5];
  const float* Wk = (const float*)d_in[6];  const float* bk = (const float*)d_in[7];
  const float* Wv = (const float*)d_in[8];  const float* bv = (const float*)d_in[9];
  const float* Wcq = (const float*)d_in[10]; const float* bcq = (const float*)d_in[11];
  const float* Wck = (const float*)d_in[12]; const float* bck = (const float*)d_in[13];
  const float* Whcq = (const float*)d_in[14]; const float* bhcq = (const float*)d_in[15];
  const float* Whck = (const float*)d_in[16]; const float* bhck = (const float*)d_in[17];
  const float* wlqc = (const float*)d_in[18]; const float* wlqq = (const float*)d_in[19];
  const float* wlkc = (const float*)d_in[20]; const float* wlkk = (const float*)d_in[21];

  constexpr size_t NHEAD = (size_t)B_ * NH_ * S_ * DH_;  // 3,145,728
  constexpr size_t NCE   = (size_t)B_ * S_ * DH_;        //   262,144
  char* w = (char*)d_ws;
  float* qf  = (float*)w;  w += NHEAD * sizeof(float);
  float* kf  = (float*)w;  w += NHEAD * sizeof(float);
  bf16*  vt  = (bf16*)w;   w += NHEAD * sizeof(bf16);
  float* hcq = (float*)w;  w += NHEAD * sizeof(float);
  float* hck = (float*)w;  w += NHEAD * sizeof(float);
  float* ceq = (float*)w;  w += NCE * sizeof(float);
  float* cek = (float*)w;  w += NCE * sizeof(float);
  bf16* cq_layer = (bf16*)w; w += NHEAD * sizeof(bf16);
  bf16* ck_layer = (bf16*)w; w += NHEAD * sizeof(bf16);

  float* ctx_out   = (float*)d_out;
  float* probs_out = ctx_out + (size_t)B_ * S_ * H_;

  // 1) Q/K/V (3 matrices fused per block)
  k_qkv<<<dim3(B_ * S_ / 16, H_ / 16), 32, 0, stream>>>(
      hidden, Wq, bq, Wk, bk, Wv, bv, qf, kf, vt);
  // 2a) head-context -> hcq/hck: 151MB tensor streamed exactly once
  k_proj64<<<dim3((size_t)B_ * NH_ * S_ / 16), 32, 0, stream>>>(
      hctx, Wcq, bcq, Wck, bck, hcq, hck);
  // 2b) context -> ceq/cek
  k_proj64<<<dim3(B_ * S_ / 16), 32, 0, stream>>>(
      ctxemb, Wcq, bcq, Wck, bck, ceq, cek);
  // 3) concat-GEMM + tanh gate + blend
  k_gate<<<dim3((size_t)B_ * NH_ * S_ / 16), 32, 0, stream>>>(
      hcq, ceq, hck, cek, Whcq, bhcq, Whck, bhck,
      wlqc, wlqq, wlkc, wlkk, qf, kf, cq_layer, ck_layer);
  // 4) fused attention
  const size_t smem = (size_t)16 * (S_ + 4) * sizeof(float)
                    + 2 * 4 * 16 * 16 * sizeof(float)
                    + 16 * 64 * sizeof(bf16);
  k_attn<<<dim3(S_ / 16, B_ * NH_), 256, smem, stream>>>(
      cq_layer, ck_layer, vt, mask, ctx_out, probs_out);
}
